// SurvivalGraphArch_12953621365188
// MI455X (gfx1250) — matrix operand order repaired and checked by reading, not compiled, not run on hardware
//
#include <hip/hip_runtime.h>
#include <hip/hip_bf16.h>

// ---------------------------------------------------------------------------
// SurvivalGraphArch pipeline for MI455X (gfx1250), wave32 + WMMA bf16.
// GEMMs with K%32==0 && N%32==0 use v_wmma_f32_16x16x32_bf16 (f32 accum),
// double-buffered LDS tiles (global loads overlap WMMA), fragments loaded as
// ds_load_b128 pairs. Row indices are clamped (not guarded) at tile load; the
// epilogue guards stores, so no EXEC-masked slow path exists.
// Tiny GEMMs (K=2 / N=2 / N=4) stay on VALU. GIN "h + agg" is fused into the
// WMMA A-tile load. All buffers come from d_ws.
// ---------------------------------------------------------------------------

typedef __attribute__((ext_vector_type(16))) __bf16 v16bf;
typedef __attribute__((ext_vector_type(8)))  __bf16 v8bf;
typedef __attribute__((ext_vector_type(4)))  __bf16 v4bf;
typedef __attribute__((ext_vector_type(8)))  float  v8f;

#define GN   50000
#define GE   800000
#define GF   1024
#define GL0  256
#define GL1  128
#define GL2  128
#define GB   8
#define GT   4

#define AST  40   // LDS row stride (bf16 elems): 80B = 20 banks, 16B aligned
#define WST  40

// ------------------------------- WMMA GEMM ---------------------------------
// C[M x Nn] = act( (A [+ A2]) @ W + bias ), A row-major [M x K], W [K x Nn].
// Block = 256 threads = 8 waves. Tile: BM=128 rows, BN=32 cols, BK=32.
// Wave w owns rows [m0+16w, m0+16w+16) and both 16-wide column tiles.
// Requires K%32==0, Nn%32==0.
template <bool HAS_A2>
__global__ __launch_bounds__(256)
void wmma_gemm_bf16(const float* __restrict__ A, const float* __restrict__ A2,
                    const float* __restrict__ W, const float* __restrict__ bias,
                    float* __restrict__ C, int M, int K, int Nn, int relu) {
    __shared__ __align__(16) __bf16 lds_a[2][128 * AST]; // A tile: [row][k]
    __shared__ __align__(16) __bf16 lds_w[2][32 * WST];  // W tile transposed [n][k]

    const int n0 = blockIdx.x * 32;
    const int m0 = blockIdx.y * 128;
    const int tid  = threadIdx.x;
    const int lane = tid & 31;
    const int wv   = tid >> 5;
    const int l16  = lane & 15;
    const int half = lane >> 4;            // 0: lanes 0-15, 1: lanes 16-31

    float4 ra[4], rb[4], rw;

    // issue all global loads for one k-tile into registers (no waits between)
    auto loadT = [&](int k0) {
        #pragma unroll
        for (int i = 0; i < 4; ++i) {
            int idx = tid + i * 256;           // 0..1023 float4 slots
            int r   = idx >> 3;                // tile row 0..127
            int kc  = (idx & 7) * 4;           // k offset 0,4,..,28
            int gm  = m0 + r;
            gm = (gm < M) ? gm : (M - 1);      // clamp: stores are guarded later
            size_t off = (size_t)gm * K + (k0 + kc);
            ra[i] = *(const float4*)&A[off];
            if (HAS_A2) rb[i] = *(const float4*)&A2[off];
            if (i == 0 && k0 + 32 < K)
                __builtin_prefetch(&A[off + 32], 0, 3);   // global_prefetch_b8
        }
        int r  = tid >> 3;                     // k row 0..31
        int c4 = (tid & 7) * 4;                // n offset 0,4,..,28
        rw = *(const float4*)&W[(size_t)(k0 + r) * Nn + (n0 + c4)];
    };

    // convert registers -> bf16 LDS tiles (buffer p)
    auto storeT = [&](int p) {
        #pragma unroll
        for (int i = 0; i < 4; ++i) {
            int idx = tid + i * 256;
            int r   = idx >> 3;
            int kc  = (idx & 7) * 4;
            float4 v = ra[i];
            if (HAS_A2) {
                v.x += rb[i].x; v.y += rb[i].y; v.z += rb[i].z; v.w += rb[i].w;
            }
            v4bf pk;
            pk[0] = (__bf16)v.x; pk[1] = (__bf16)v.y;
            pk[2] = (__bf16)v.z; pk[3] = (__bf16)v.w;
            *(v4bf*)&lds_a[p][r * AST + kc] = pk;
        }
        int r  = tid >> 3;
        int c4 = (tid & 7) * 4;
        lds_w[p][(c4 + 0) * WST + r] = (__bf16)rw.x;
        lds_w[p][(c4 + 1) * WST + r] = (__bf16)rw.y;
        lds_w[p][(c4 + 2) * WST + r] = (__bf16)rw.z;
        lds_w[p][(c4 + 3) * WST + r] = (__bf16)rw.w;
    };

    v8f acc0 = {}, acc1 = {};

    // prologue: stage tile 0
    loadT(0);
    storeT(0);
    __syncthreads();

    int p = 0;
    for (int k0 = 0; k0 < K; k0 += 32) {
        const bool more = (k0 + 32) < K;
        if (more) loadT(k0 + 32);              // overlap with WMMA below

        // ---- fragments (wave32 16-bit, contiguous k order): 2x b128 loads each ----
        // A 16x32: lane row = 16*wv + l16; K = alo+e (alo = 16*half)
        const int arow = (16 * wv + l16) * AST;
        const int alo  = half * 16;
        v8bf a0 = *(const v8bf*)&lds_a[p][arow + alo];
        v8bf a1 = *(const v8bf*)&lds_a[p][arow + alo + 8];
        v16bf af = __builtin_shufflevector(a0, a1,
                       0,1,2,3,4,5,6,7,8,9,10,11,12,13,14,15);
        // B 32x16: lane col = l16 (tile0) / 16+l16 (tile1); K = bko+e
        const int bko = half * 16;
        v8bf b0 = *(const v8bf*)&lds_w[p][l16 * WST + bko];
        v8bf b1 = *(const v8bf*)&lds_w[p][l16 * WST + bko + 8];
        v16bf bf0 = __builtin_shufflevector(b0, b1,
                        0,1,2,3,4,5,6,7,8,9,10,11,12,13,14,15);
        v8bf b2 = *(const v8bf*)&lds_w[p][(16 + l16) * WST + bko];
        v8bf b3 = *(const v8bf*)&lds_w[p][(16 + l16) * WST + bko + 8];
        v16bf bf1 = __builtin_shufflevector(b2, b3,
                        0,1,2,3,4,5,6,7,8,9,10,11,12,13,14,15);

        acc0 = __builtin_amdgcn_wmma_f32_16x16x32_bf16(
                   false, af, false, bf0, (short)0, acc0, false, false);
        acc1 = __builtin_amdgcn_wmma_f32_16x16x32_bf16(
                   false, af, false, bf1, (short)0, acc1, false, false);

        if (more) storeT(p ^ 1);               // fill the other buffer
        __syncthreads();
        p ^= 1;
    }

    // ---- epilogue: C/D layout -> VGPR r holds row rbase+r, col = l16 ----
    const int col0  = n0 + l16;
    const int col1  = col0 + 16;
    const int rbase = m0 + 16 * wv + half * 8;
    const float bv0 = bias ? bias[col0] : 0.0f;
    const float bv1 = bias ? bias[col1] : 0.0f;
    #pragma unroll
    for (int r = 0; r < 8; ++r) {
        int row = rbase + r;
        if (row < M) {
            float v0 = acc0[r] + bv0;
            float v1 = acc1[r] + bv1;
            if (relu) { v0 = fmaxf(v0, 0.0f); v1 = fmaxf(v1, 0.0f); }
            C[(size_t)row * Nn + col0] = v0;
            C[(size_t)row * Nn + col1] = v1;
        }
    }
}

// ----------------------------- scalar helpers ------------------------------
__global__ void zero_f32(float* p, int n) {
    int i = blockIdx.x * blockDim.x + threadIdx.x;
    if (i < n) p[i] = 0.0f;
}

// generic small GEMM: C = act(A@W + b), one thread per output element
__global__ void small_gemm(const float* __restrict__ A, const float* __restrict__ W,
                           const float* __restrict__ bias, float* __restrict__ C,
                           int M, int K, int Nn, int relu) {
    int tid = blockIdx.x * blockDim.x + threadIdx.x;
    if (tid >= M * Nn) return;
    int m = tid / Nn, n = tid % Nn;
    float s = bias[n];
    for (int k = 0; k < K; ++k) s += A[(size_t)m * K + k] * W[(size_t)k * Nn + n];
    if (relu) s = fmaxf(s, 0.0f);
    C[tid] = s;
}

// coalesced per-column partial sums: thread t owns column t for a row chunk
#define BN_ROWS 512
__global__ __launch_bounds__(256)
void bn_partial(const float* __restrict__ h, float* __restrict__ sums,
                float* __restrict__ sqs, int M, int C) {
    int c  = threadIdx.x;                 // C == 256 == blockDim.x
    int r0 = blockIdx.x * BN_ROWS;
    int r1 = min(r0 + BN_ROWS, M);
    float s = 0.0f, q = 0.0f;
    for (int r = r0; r < r1; ++r) {
        float v = h[(size_t)r * C + c];   // coalesced across the wave
        s += v; q += v * v;
    }
    atomicAdd(&sums[c], s);
    atomicAdd(&sqs[c], q);
}

__global__ void bn_finalize(const float* __restrict__ sums, const float* __restrict__ sqs,
                            float* __restrict__ mean, float* __restrict__ rstd, int M) {
    int c = threadIdx.x;
    float mu  = sums[c] / (float)M;
    float var = sqs[c] / (float)M - mu * mu;
    mean[c] = mu;
    rstd[c] = rsqrtf(var + 1e-5f);
}

__global__ void bn_relu(float* __restrict__ h, const float* __restrict__ mean,
                        const float* __restrict__ rstd, const float* __restrict__ g,
                        const float* __restrict__ b, int total, int C) {
    int i = blockIdx.x * blockDim.x + threadIdx.x;
    if (i >= total) return;
    int c = i % C;
    float v = (h[i] - mean[c]) * rstd[c] * g[c] + b[c];
    h[i] = fmaxf(v, 0.0f);
}

// agg[dst[e], :] += h[src[e], :] over all edges (D floats per node)
__global__ void edge_scatter_add(const float* __restrict__ h,
                                 const int* __restrict__ src,
                                 const int* __restrict__ dst,
                                 float* __restrict__ agg, int E, int D) {
    int tid = blockIdx.x * blockDim.x + threadIdx.x;
    if (tid >= E * D) return;
    int e = tid / D, f = tid % D;
    atomicAdd(&agg[(size_t)dst[e] * D + f], h[(size_t)src[e] * D + f]);
}

// enc -> loc, clipped logvar, gate = loc + exp(0.5*logvar)*eps
__global__ void gate_kernel(const float* __restrict__ enc, const float* __restrict__ eps,
                            float* __restrict__ gate, float* __restrict__ out_loc,
                            float* __restrict__ out_lv, int M) {
    int i = blockIdx.x * blockDim.x + threadIdx.x;
    if (i >= M) return;
    float e0 = enc[2 * i], e1 = enc[2 * i + 1];
    float lv = fminf(fmaxf(e1, -20.0f), 20.0f);
    out_loc[i] = e0;
    out_lv[i]  = lv;
    gate[i]    = e0 + expf(0.5f * lv) * eps[i];
}

// num[batch[i], :] += feat[i, :] * gate[i];  cnt[batch[i]] += 1
__global__ void pool_scatter(const float* __restrict__ feat, const float* __restrict__ gate,
                             const int* __restrict__ batch, float* __restrict__ num,
                             float* __restrict__ cnt, int M, int D) {
    int tid = blockIdx.x * blockDim.x + threadIdx.x;
    if (tid >= M * D) return;
    int i = tid / D, f = tid % D;
    int b = batch[i];
    atomicAdd(&num[(size_t)b * D + f], feat[tid] * gate[i]);
    if (f == 0) atomicAdd(&cnt[b], 1.0f);
}

// output[b,t] = sum_f (num[b,f]/max(cnt[b],1)) * Wc[f,t] + bc[t]
__global__ void final_out(const float* __restrict__ num, const float* __restrict__ cnt,
                          const float* __restrict__ Wc, const float* __restrict__ bc,
                          float* __restrict__ out, int D) {
    int tid = threadIdx.x;            // 32 threads: b = tid/4, t = tid%4
    if (tid >= GB * GT) return;
    int b = tid >> 2, t = tid & 3;
    float inv = 1.0f / fmaxf(cnt[b], 1.0f);
    float s = bc[t];
    for (int f = 0; f < D; ++f) s += num[(size_t)b * D + f] * inv * Wc[(size_t)f * GT + t];
    out[tid] = s;
}

// ------------------------------- launcher ----------------------------------
static inline dim3 wgrid(int M, int Nn) { return dim3(Nn / 32, (M + 127) / 128, 1); }

extern "C" void kernel_launch(void* const* d_in, const int* in_sizes, int n_in,
                              void* d_out, int out_size, void* d_ws, size_t ws_size,
                              hipStream_t stream) {
    const float* x    = (const float*)d_in[0];
    const float* eps  = (const float*)d_in[1];
    const float* Wh   = (const float*)d_in[2];
    const float* bh   = (const float*)d_in[3];
    const float* bn_g = (const float*)d_in[4];
    const float* bn_b = (const float*)d_in[5];
    const float* sW0  = (const float*)d_in[6];
    const float* sb0  = (const float*)d_in[7];
    const float* sWg  = (const float*)d_in[8];
    const float* sbg  = (const float*)d_in[9];
    const float* sWl  = (const float*)d_in[10];
    const float* sbl  = (const float*)d_in[11];
    const float* sWt  = (const float*)d_in[12];
    const float* sbt  = (const float*)d_in[13];
    const float* iW0  = (const float*)d_in[14];
    const float* ib0  = (const float*)d_in[15];
    const float* iWg1 = (const float*)d_in[16];
    const float* ibg1 = (const float*)d_in[17];
    const float* iWl1 = (const float*)d_in[18];
    const float* ibl1 = (const float*)d_in[19];
    const float* iWg2 = (const float*)d_in[20];
    const float* ibg2 = (const float*)d_in[21];
    const float* iWl2 = (const float*)d_in[22];
    const float* ibl2 = (const float*)d_in[23];
    const float* iWt  = (const float*)d_in[24];
    const float* ibt  = (const float*)d_in[25];
    const float* Wc   = (const float*)d_in[26];
    const float* bc   = (const float*)d_in[27];
    const int*   ei   = (const int*)d_in[28];
    const int*   batch= (const int*)d_in[29];
    const int* src = ei;
    const int* dst = ei + GE;

    float* out        = (float*)d_out;          // [32] output
    float* out_loc    = out + GB * GT;          // [N] loc
    float* out_lv     = out_loc + GN;           // [N] logvar
    float* out_decode = out_lv + GN;            // [N*256] decode

    // workspace carve-up (floats)
    float* ws   = (float*)d_ws;
    float* feat = ws;                 ws += (size_t)GN * GL0;   // N x 256
    float* h    = ws;                 ws += (size_t)GN * GL1;   // N x 128
    float* agg  = ws;                 ws += (size_t)GN * GL0;   // N x <=256
    float* tbuf = ws;                 ws += (size_t)GN * GL0;   // N x <=256
    float* g    = ws;                 ws += (size_t)GN * GL0;   // N x <=256
    float* enc  = ws;                 ws += (size_t)GN * 2;     // N x 2
    float* gate = ws;                 ws += (size_t)GN;         // N
    float* mean = ws;                 ws += GL0;
    float* rstd = ws;                 ws += GL0;
    float* bnsum= ws;                 ws += GL0;
    float* bnsq = ws;                 ws += GL0;
    float* num  = ws;                 ws += (size_t)GB * GL0;   // 8 x 256
    float* cnt  = ws;                 ws += GB;                 // 8

    const int TPB = 256;
    auto blks = [](int n) { return (n + 255) / 256; };

    // 1) pre-BN: feat_raw = x @ Wh + bh
    wmma_gemm_bf16<false><<<wgrid(GN, GL0), TPB, 0, stream>>>(x, nullptr, Wh, bh, feat,
                                                              GN, GF, GL0, 0);
    // 2-3) batchnorm + relu (in place), coalesced two-stage stats
    zero_f32<<<blks(2 * GL0), TPB, 0, stream>>>(bnsum, 2 * GL0);
    bn_partial<<<(GN + BN_ROWS - 1) / BN_ROWS, TPB, 0, stream>>>(feat, bnsum, bnsq,
                                                                 GN, GL0);
    bn_finalize<<<1, GL0, 0, stream>>>(bnsum, bnsq, mean, rstd, GN);
    bn_relu<<<blks(GN * GL0), TPB, 0, stream>>>(feat, mean, rstd, bn_g, bn_b,
                                                GN * GL0, GL0);
    // 4) h = relu(feat @ sW0 + sb0)
    wmma_gemm_bf16<false><<<wgrid(GN, GL1), TPB, 0, stream>>>(feat, nullptr, sW0, sb0, h,
                                                              GN, GL0, GL1, 1);
    // 5-7) GIN (shared): h = relu((h+agg)@sWg+sbg) @ sWl + sbl
    zero_f32<<<blks(GN * GL1), TPB, 0, stream>>>(agg, GN * GL1);
    edge_scatter_add<<<blks(GE * GL1), TPB, 0, stream>>>(h, src, dst, agg, GE, GL1);
    wmma_gemm_bf16<true><<<wgrid(GN, GL2), TPB, 0, stream>>>(h, agg, sWg, sbg, tbuf,
                                                             GN, GL1, GL2, 1);
    wmma_gemm_bf16<false><<<wgrid(GN, GL2), TPB, 0, stream>>>(tbuf, nullptr, sWl, sbl, h,
                                                              GN, GL2, GL2, 0);
    // 8) enc = h @ sWt + sbt  (N=2: VALU)
    small_gemm<<<blks(GN * 2), TPB, 0, stream>>>(h, sWt, sbt, enc, GN, GL2, 2, 0);
    // 9) loc / logvar / gate
    gate_kernel<<<blks(GN), TPB, 0, stream>>>(enc, eps, gate, out_loc, out_lv, GN);
    // 10) g = relu(enc @ iW0 + ib0)  (K=2: VALU)
    small_gemm<<<blks(GN * GL2), TPB, 0, stream>>>(enc, iW0, ib0, g, GN, 2, GL2, 1);
    // 11-13) GIN i1 (128 -> 128 -> 128)
    zero_f32<<<blks(GN * GL2), TPB, 0, stream>>>(agg, GN * GL2);
    edge_scatter_add<<<blks(GE * GL2), TPB, 0, stream>>>(g, src, dst, agg, GE, GL2);
    wmma_gemm_bf16<true><<<wgrid(GN, GL1), TPB, 0, stream>>>(g, agg, iWg1, ibg1, tbuf,
                                                             GN, GL2, GL1, 1);
    wmma_gemm_bf16<false><<<wgrid(GN, GL1), TPB, 0, stream>>>(tbuf, nullptr, iWl1, ibl1, g,
                                                              GN, GL1, GL1, 0);
    // 14-16) GIN i2 (128 -> 256 -> 256)
    zero_f32<<<blks(GN * GL1), TPB, 0, stream>>>(agg, GN * GL1);
    edge_scatter_add<<<blks(GE * GL1), TPB, 0, stream>>>(g, src, dst, agg, GE, GL1);
    wmma_gemm_bf16<true><<<wgrid(GN, GL0), TPB, 0, stream>>>(g, agg, iWg2, ibg2, tbuf,
                                                             GN, GL1, GL0, 1);
    wmma_gemm_bf16<false><<<wgrid(GN, GL0), TPB, 0, stream>>>(tbuf, nullptr, iWl2, ibl2, g,
                                                              GN, GL0, GL0, 0);
    // 17) decode = g @ iWt + ibt  -> directly into d_out
    wmma_gemm_bf16<false><<<wgrid(GN, GL0), TPB, 0, stream>>>(g, nullptr, iWt, ibt,
                                                              out_decode, GN, GL0, GL0, 0);
    // 18-20) masked mean pool + classifier head
    zero_f32<<<blks(GB * GL0 + GB), TPB, 0, stream>>>(num, GB * GL0 + GB);
    pool_scatter<<<blks(GN * GL0), TPB, 0, stream>>>(feat, gate, batch, num, cnt,
                                                     GN, GL0);
    final_out<<<1, 32, 0, stream>>>(num, cnt, Wc, bc, out, GL0);
}
